// GAT_56152402428487
// MI455X (gfx1250) — matrix operand order repaired and checked by reading, not compiled, not run on hardware
//
#include <hip/hip_runtime.h>

#define NN 50000
#define NPAD 50016          // 50016 = 32 * 1563: M padded for 2x4 register-blocked WMMA
#define NE 800000
#define IN_C 128
#define HID_C 32
#define OUTC 64
#define HEADS 8
#define NEG_SLOPE 0.2f
#define EPS_F 1e-16f

#define CEILDIV(a, b) (((a) + (b) - 1) / (b))

typedef __attribute__((ext_vector_type(16))) _Float16 v16h;
typedef __attribute__((ext_vector_type(8)))  float    v8f;
typedef __attribute__((ext_vector_type(4)))  unsigned u32x4;
typedef __attribute__((ext_vector_type(4)))  int      i32x4;
typedef __attribute__((ext_vector_type(8)))  int      i32x8;

// ---------- helpers: order-preserving float <-> uint for atomicMax ----------
__device__ __forceinline__ unsigned f2ord(float f) {
  unsigned u = __float_as_uint(f);
  return (u & 0x80000000u) ? ~u : (u | 0x80000000u);
}
__device__ __forceinline__ float ord2f(unsigned k) {
  unsigned u = (k & 0x80000000u) ? (k & 0x7FFFFFFFu) : ~k;
  return __uint_as_float(u);
}
#define ORD_NEG_INF 0x007FFFFFu  // f2ord(-inf)

// ---------- TDM: stage a [rows, K]-f16 row-major panel into LDS ----------
// D# per CDNA5 ISA. 2D tile, fully contiguous
// (tile_dim0 == tensor_dim0_stride == K), data_size = 2 bytes.
__device__ __forceinline__ void tdm_stage_panel(const _Float16* gsrc, void* lds_dst,
                                                int K_elems, int rows) {
#if __has_builtin(__builtin_amdgcn_tensor_load_to_lds)
  unsigned long long ga = (unsigned long long)gsrc;
  unsigned lds_addr = (unsigned)(unsigned long long)lds_dst;  // addr[31:0] = LDS offset
  u32x4 g0;
  g0[0] = 1u;                                     // count=1, user mode, no gather
  g0[1] = lds_addr;                               // lds_addr
  g0[2] = (unsigned)ga;                           // global_addr[31:0]
  g0[3] = (unsigned)((ga >> 32) & 0x01FFFFFFull)  // global_addr[56:32]
          | 0x80000000u;                          // type=2 ("image") in bits 127:126
  i32x8 g1;
  g1[0] = (int)(1u << 16);                        // data_size=1 (2 bytes); wg_mask=0
  g1[1] = (int)((unsigned)K_elems << 16);         // tensor_dim0[15:0] @ bits 63:48
  g1[2] = (int)((((unsigned)K_elems >> 16) & 0xFFFFu) |
                ((unsigned)rows << 16));          // dim0[31:16] | tensor_dim1[15:0]
  g1[3] = (int)((((unsigned)rows >> 16) & 0xFFFFu) |
                ((unsigned)K_elems << 16));       // dim1[31:16] | tile_dim0
  g1[4] = (int)(unsigned)rows;                    // tile_dim1 (tile_dim2 = 0)
  g1[5] = (int)(unsigned)K_elems;                 // tensor_dim0_stride[31:0]
  g1[6] = 0;                                      // stride0[47:32] | stride1[15:0]
  g1[7] = 0;
#if __clang_major__ >= 23
  __builtin_amdgcn_tensor_load_to_lds(g0, g1, (i32x4){0, 0, 0, 0},
                                      (i32x4){0, 0, 0, 0},
                                      (i32x8){0, 0, 0, 0, 0, 0, 0, 0}, 0);
#else
  __builtin_amdgcn_tensor_load_to_lds(g0, g1, (i32x4){0, 0, 0, 0},
                                      (i32x4){0, 0, 0, 0}, 0);
#endif
  __builtin_amdgcn_s_wait_tensorcnt(0);
#else
  // Fallback: cooperative wave copy (one wave), 16B chunks.
  const float4* s = (const float4*)gsrc;
  float4* d = (float4*)lds_dst;
  int n16 = (K_elems * rows) >> 3;  // halfs -> float4 chunks
  for (int i = threadIdx.x & 31; i < n16; i += 32) d[i] = s[i];
#endif
}

// ---------- conversion kernels ----------
__global__ void k_cvt_f16(const float* __restrict__ in, _Float16* __restrict__ out, int n) {
  int i = blockIdx.x * blockDim.x + threadIdx.x;
  if (i < n) out[i] = (_Float16)in[i];
}

// W [K, Nc] f32 row-major  ->  WT [Nc, K] f16 row-major
__global__ void k_cvt_transpose_f16(const float* __restrict__ W, _Float16* __restrict__ WT,
                                    int K, int Nc) {
  int i = blockIdx.x * blockDim.x + threadIdx.x;
  if (i >= K * Nc) return;
  int k = i / Nc, c = i - k * Nc;
  WT[(size_t)c * K + k] = (_Float16)W[i];
}

__global__ void k_fill_u32(unsigned* __restrict__ p, unsigned v, int n) {
  int i = blockIdx.x * blockDim.x + threadIdx.x;
  if (i < n) p[i] = v;
}

// ---------- register-blocked WMMA GEMM with TDM-staged B panel ----------
// C[M,Nc] = A[M,K] * BT[Nc,K]^T (f16 in, f32 out).
// The whole BT panel (Nc*K halfs, <= 64KB) is DMA'd to LDS once per block by
// the Tensor Data Mover; waves read B fragments from LDS (ds_load_b128) and
// A fragments from global (4 b128 loads per K-step), 8 v_wmma per K-step.
// Requires: M % 32 == 0, Nc % 64 == 0, K % 32 == 0, Nc*K <= 32768.
__global__ void k_wmma_gemm2x4(const _Float16* __restrict__ A,
                               const _Float16* __restrict__ BT,
                               float* __restrict__ C,
                               int M, int K, int Nc) {
  __shared__ __align__(16) _Float16 sBT[32768];  // 64KB panel

  const int lane = threadIdx.x & 31;
  const int wave = blockIdx.x * (blockDim.x >> 5) + (threadIdx.x >> 5);

  if (threadIdx.x < 32)  // wave 0 issues the tensor DMA and waits on TENSORcnt
    tdm_stage_panel(BT, sBT, K, Nc);
  __syncthreads();

  const int stripsN = Nc >> 6;            // 64-wide N strips
  const int tm = wave / stripsN;          // 32-row block index
  const int tn = wave - tm * stripsN;
  if (tm * 32 >= M) return;  // wave-uniform guard (EXEC stays all-ones for WMMA)

  const int  l15 = lane & 15;
  const bool hi  = lane >= 16;
  const int  ka  = hi ? 8 : 0;    // A frag: hi lanes carry K+8 / K+24 halves
  const int  kb  = hi ? 8 : 0;    // B frag: hi lanes carry K+8 / K+24 halves

  const _Float16* Arow0 = A + (size_t)(tm * 32 + l15) * K;
  const _Float16* Arow1 = Arow0 + (size_t)16 * K;
  const _Float16* Brow  = sBT + (size_t)(tn * 64 + l15) * K;

  union Frag { v16h v; float4 f[2]; };
  v8f acc[2][4] = {};
  for (int k0 = 0; k0 < K; k0 += 32) {
    Frag a0, a1, b[4];
    a0.f[0] = *(const float4*)(Arow0 + k0 + ka);
    a0.f[1] = *(const float4*)(Arow0 + k0 + ka + 16);
    a1.f[0] = *(const float4*)(Arow1 + k0 + ka);
    a1.f[1] = *(const float4*)(Arow1 + k0 + ka + 16);
#pragma unroll
    for (int j = 0; j < 4; ++j) {
      const _Float16* Br = Brow + (size_t)(j * 16) * K;
      b[j].f[0] = *(const float4*)(Br + k0 + kb);
      b[j].f[1] = *(const float4*)(Br + k0 + kb + 16);
    }
#pragma unroll
    for (int j = 0; j < 4; ++j) {
      acc[0][j] = __builtin_amdgcn_wmma_f32_16x16x32_f16(false, a0.v, false, b[j].v,
                                                         (short)0, acc[0][j], false, false);
      acc[1][j] = __builtin_amdgcn_wmma_f32_16x16x32_f16(false, a1.v, false, b[j].v,
                                                         (short)0, acc[1][j], false, false);
    }
  }
  // D layout: VGPR r -> M = base + r + (hi?8:0), N = tile_col + l15
  const int rb = tm * 32 + (hi ? 8 : 0);
#pragma unroll
  for (int i = 0; i < 2; ++i) {
#pragma unroll
    for (int j = 0; j < 4; ++j) {
      float* Cp = C + (size_t)(rb + i * 16) * Nc + tn * 64 + j * 16 + l15;
#pragma unroll
      for (int r = 0; r < 8; ++r)
        Cp[(size_t)r * Nc] = acc[i][j][r];
    }
  }
}

// ---------- per-node attention logits: a_src/a_dst [N,H] ----------
__global__ void k_node_att(const float* __restrict__ h, const float* __restrict__ att_src,
                           const float* __restrict__ att_dst,
                           float* __restrict__ a_src, float* __restrict__ a_dst,
                           int N, int H, int C) {
  int i = blockIdx.x * blockDim.x + threadIdx.x;
  if (i >= N * H) return;
  int n = i / H, hh = i - n * H;
  const float* hrow = h + (size_t)n * H * C + (size_t)hh * C;
  const float* as = att_src + hh * C;
  const float* ad = att_dst + hh * C;
  float s = 0.f, d = 0.f;
  for (int c = 0; c < C; ++c) { float v = hrow[c]; s += v * as[c]; d += v * ad[c]; }
  a_src[i] = s;
  a_dst[i] = d;
}

// ---------- edge pass 1: e = leakyrelu(a_src[src]+a_dst[dst]); segment max ----------
__global__ void k_edge_max(const int* __restrict__ src, const int* __restrict__ dst,
                           const float* __restrict__ a_src, const float* __restrict__ a_dst,
                           float* __restrict__ e_out, unsigned* __restrict__ m_ord,
                           int E, int H) {
  int i = blockIdx.x * blockDim.x + threadIdx.x;
  if (i >= E * H) return;
  int e = i / H, hh = i - e * H;
  int s = src[e], d = dst[e];
  float v = a_src[s * H + hh] + a_dst[d * H + hh];
  v = v > 0.f ? v : NEG_SLOPE * v;
  e_out[i] = v;
  atomicMax(m_ord + d * H + hh, f2ord(v));
}

// ---------- edge pass 2: ee = exp(e - m[dst]); segment sum denom ----------
__global__ void k_edge_exp(const int* __restrict__ dst, float* __restrict__ e_buf,
                           const unsigned* __restrict__ m_ord, float* __restrict__ denom,
                           int E, int H) {
  int i = blockIdx.x * blockDim.x + threadIdx.x;
  if (i >= E * H) return;
  int e = i / H, hh = i - e * H;
  int d = dst[e];
  float m = ord2f(m_ord[d * H + hh]);
  if (!(m >= -3.4e38f)) m = 0.f;  // -inf (no incoming edges) -> 0, matching reference
  float ee = __expf(e_buf[i] - m);
  e_buf[i] = ee;
  atomicAdd(denom + d * H + hh, ee);
}

// ---------- aggregation: out[dst] += h[src] * alpha ----------
// One wave per 4 edges; prefetch the next edge's source row (global_prefetch_b8)
// while streaming atomics for the current one.
#define AGG_EPW 4
__global__ void k_edge_agg(const int* __restrict__ src, const int* __restrict__ dst,
                           const float* __restrict__ h, const float* __restrict__ ee,
                           const float* __restrict__ denom, float* __restrict__ out,
                           int E, int H, int C) {
  int wave = blockIdx.x * (blockDim.x >> 5) + (threadIdx.x >> 5);
  int lane = threadIdx.x & 31;
  int HC = H * C;
  int e0 = wave * AGG_EPW;
  int e1 = e0 + AGG_EPW < E ? e0 + AGG_EPW : E;
  if (e0 >= E) return;
  int fpl = HC >> 5;  // floats per lane in one row
  for (int e = e0; e < e1; ++e) {
    if (e + 1 < e1) {  // prefetch next source row into WGP$/L2
      const float* nxt = h + (size_t)src[e + 1] * HC + lane * fpl;
      __builtin_prefetch(nxt, 0, 0);
    }
    int s = src[e], d = dst[e];
    const float* hrow = h + (size_t)s * HC;
    float* orow = out + (size_t)d * HC;
    for (int idx = lane; idx < HC; idx += 32) {
      int hh = idx / C;
      float alpha = ee[(size_t)e * H + hh] / (denom[d * H + hh] + EPS_F);
      atomicAdd(orow + idx, hrow[idx] * alpha);
    }
  }
}

// ---------- bias + ELU + convert to f16 (feeds GEMM2) ----------
__global__ void k_bias_elu_f16(const float* __restrict__ in, const float* __restrict__ bias,
                               _Float16* __restrict__ out, int total, int Cc) {
  int i = blockIdx.x * blockDim.x + threadIdx.x;
  if (i >= total) return;
  int c = i % Cc;
  float v = in[i] + bias[c];
  v = v > 0.f ? v : (__expf(v) - 1.f);
  out[i] = (_Float16)v;
}

// ---------- final bias add, in place ----------
__global__ void k_bias_add(float* __restrict__ out, const float* __restrict__ bias,
                           int total, int Cc) {
  int i = blockIdx.x * blockDim.x + threadIdx.x;
  if (i >= total) return;
  out[i] += bias[i % Cc];
}

extern "C" void kernel_launch(void* const* d_in, const int* in_sizes, int n_in,
                              void* d_out, int out_size, void* d_ws, size_t ws_size,
                              hipStream_t stream) {
  const float* x     = (const float*)d_in[0];
  const int*   ei    = (const int*)d_in[1];
  const float* W1    = (const float*)d_in[2];
  const float* attS1 = (const float*)d_in[3];
  const float* attD1 = (const float*)d_in[4];
  const float* bias1 = (const float*)d_in[5];
  const float* W2    = (const float*)d_in[6];
  const float* attS2 = (const float*)d_in[7];
  const float* attD2 = (const float*)d_in[8];
  const float* bias2 = (const float*)d_in[9];
  const int* src = ei;
  const int* dst = ei + NE;
  float* out = (float*)d_out;

  // ---- carve workspace (256B-aligned regions) ----
  char*  ws  = (char*)d_ws;
  size_t off = 0;
  auto carve = [&](size_t bytes) -> char* {
    char* p = ws + off;
    off = (off + bytes + 255) & ~(size_t)255;
    return p;
  };
  size_t ee1_bytes = (size_t)NE * 8 * 4;                 // ee1 [E,8] f32
  size_t helu_bytes = (size_t)NPAD * 256 * 2;            // helu [NPAD,256] f16 (reuses ee1)
  size_t big = ee1_bytes > helu_bytes ? ee1_bytes : helu_bytes;

  _Float16* xf16 = (_Float16*)carve((size_t)NPAD * IN_C * 2);      // padded rows: don't-care
  _Float16* wt1  = (_Float16*)carve((size_t)(HEADS * HID_C) * IN_C * 2);
  _Float16* wt2  = (_Float16*)carve((size_t)OUTC * (HEADS * HID_C) * 2);
  float* h1   = (float*)carve((size_t)NPAD * 256 * 4);   // reused as h2 [NPAD,64]
  float* out1 = (float*)carve((size_t)NN * 256 * 4);
  float* ee1  = (float*)carve(big);                      // reused as helu
  float* as1  = (float*)carve((size_t)NN * 8 * 4);       // reused as as2
  float* ad1  = (float*)carve((size_t)NN * 8 * 4);       // reused as ad2
  unsigned* m1 = (unsigned*)carve((size_t)NN * 8 * 4);   // reused as m2
  float* dn1  = (float*)carve((size_t)NN * 8 * 4);       // reused as dn2
  float* ee2  = (float*)carve((size_t)NE * 4);
  _Float16* helu = (_Float16*)ee1;
  float* h2 = h1;
  float* as2 = as1; float* ad2 = ad1;
  unsigned* m2 = m1; float* dn2 = dn1;

  const int T = 256;  // 8 wave32 per block

  // ===== layer 1 =====
  k_cvt_f16<<<CEILDIV(NN * IN_C, T), T, 0, stream>>>(x, xf16, NN * IN_C);
  k_cvt_transpose_f16<<<CEILDIV(IN_C * 256, T), T, 0, stream>>>(W1, wt1, IN_C, 256);
  k_cvt_transpose_f16<<<CEILDIV(256 * OUTC, T), T, 0, stream>>>(W2, wt2, 256, OUTC);

  // GEMM1: [NPAD,128]x[128,256] -> h1; (NPAD/32)*(256/64)=6252 waves
  k_wmma_gemm2x4<<<CEILDIV((NPAD / 32) * (256 / 64), 8), T, 0, stream>>>(
      xf16, wt1, h1, NPAD, IN_C, 256);

  k_node_att<<<CEILDIV(NN * HEADS, T), T, 0, stream>>>(h1, attS1, attD1, as1, ad1,
                                                       NN, HEADS, HID_C);
  k_fill_u32<<<CEILDIV(NN * HEADS, T), T, 0, stream>>>(m1, ORD_NEG_INF, NN * HEADS);
  k_fill_u32<<<CEILDIV(NN * HEADS, T), T, 0, stream>>>((unsigned*)dn1, 0u, NN * HEADS);
  k_fill_u32<<<CEILDIV(NN * 256, T), T, 0, stream>>>((unsigned*)out1, 0u, NN * 256);

  k_edge_max<<<CEILDIV(NE * HEADS, T), T, 0, stream>>>(src, dst, as1, ad1, ee1, m1, NE, HEADS);
  k_edge_exp<<<CEILDIV(NE * HEADS, T), T, 0, stream>>>(dst, ee1, m1, dn1, NE, HEADS);
  k_edge_agg<<<CEILDIV(NE, AGG_EPW * 8), T, 0, stream>>>(src, dst, h1, ee1, dn1, out1,
                                                         NE, HEADS, HID_C);

  k_bias_elu_f16<<<CEILDIV(NN * 256, T), T, 0, stream>>>(out1, bias1, helu, NN * 256, 256);

  // ===== layer 2 =====
  // GEMM2: [NPAD,256]x[256,64] -> h2; (NPAD/32)*(64/64)=1563 waves
  k_wmma_gemm2x4<<<CEILDIV(NPAD / 32, 8), T, 0, stream>>>(helu, wt2, h2, NPAD, 256, OUTC);

  k_node_att<<<CEILDIV(NN, T), T, 0, stream>>>(h2, attS2, attD2, as2, ad2, NN, 1, OUTC);
  k_fill_u32<<<CEILDIV(NN, T), T, 0, stream>>>(m2, ORD_NEG_INF, NN);
  k_fill_u32<<<CEILDIV(NN, T), T, 0, stream>>>((unsigned*)dn2, 0u, NN);
  k_fill_u32<<<CEILDIV(NN * OUTC, T), T, 0, stream>>>((unsigned*)out, 0u, NN * OUTC);

  k_edge_max<<<CEILDIV(NE, T), T, 0, stream>>>(src, dst, as2, ad2, ee2, m2, NE, 1);
  k_edge_exp<<<CEILDIV(NE, T), T, 0, stream>>>(dst, ee2, m2, dn2, NE, 1);
  k_edge_agg<<<CEILDIV(NE, AGG_EPW * 8), T, 0, stream>>>(src, dst, h2, ee2, dn2, out,
                                                         NE, 1, OUTC);

  k_bias_add<<<CEILDIV(NN * OUTC, T), T, 0, stream>>>(out, bias2, NN * OUTC, OUTC);
}
